// Qwen2MoeSparseMoeBlock_16587163697467
// MI455X (gfx1250) — matrix operand order repaired and checked by reading, not compiled, not run on hardware
//
#include <hip/hip_runtime.h>

#define T_TOK 1024
#define H_DIM 2048
#define E_EXP 8
#define TOPK  4
#define I_DIM 1408
#define IS_DIM 5632

#define MT 128   // macro tile M
#define NT 64    // macro tile N
#define KT 32    // K step (one bf16 WMMA)
#define LDP 40   // padded LDS row stride (bf16 elems): 80B rows, keeps 16B alignment

typedef __attribute__((ext_vector_type(16))) __bf16 v16bf;
typedef __attribute__((ext_vector_type(8)))  __bf16 v8bf;
typedef __attribute__((ext_vector_type(8)))  float  v8f;
typedef __attribute__((ext_vector_type(4)))  int    i32x4;

// round-to-nearest (half-up) bf16; 1 VALU per float + 1 v_perm per pair
__device__ __forceinline__ unsigned int pack2(float a, float b) {
  return __builtin_amdgcn_perm(__float_as_uint(b) + 0x8000u,
                               __float_as_uint(a) + 0x8000u, 0x07060302u);
}
__device__ __forceinline__ unsigned short f2bf(float f) {
  return (unsigned short)((__float_as_uint(f) + 0x8000u) >> 16);
}
__device__ __forceinline__ v8f v8f_zero() {
  v8f z;
#pragma unroll
  for (int i = 0; i < 8; ++i) z[i] = 0.f;
  return z;
}

// ---------------- gfx1250 async global->LDS copy (ASYNCcnt path), with sync fallback.
#if defined(__HIP_DEVICE_COMPILE__) && \
    __has_builtin(__builtin_amdgcn_global_load_async_to_lds_b128) && \
    __has_builtin(__builtin_amdgcn_s_wait_asynccnt)
#define USE_ASYNC_LDS 1
#else
#define USE_ASYNC_LDS 0
#endif

__device__ __forceinline__ void async_cp16(const unsigned short* g, unsigned short* l) {
#if USE_ASYNC_LDS
  __builtin_amdgcn_global_load_async_to_lds_b128(
      (__attribute__((address_space(1))) i32x4*)g,
      (__attribute__((address_space(3))) i32x4*)l, 0, 0);
#else
  *(uint4*)l = *(const uint4*)g;
#endif
}
__device__ __forceinline__ void async_fence() {
#if USE_ASYNC_LDS
  __builtin_amdgcn_s_wait_asynccnt(0);
#endif
}

// ---------------- Router: softmax(fp32) -> top-4 -> renorm; plus sigmoid shared gate.
__global__ __launch_bounds__(256) void router_kernel(
    const float* __restrict__ x, const float* __restrict__ gw,
    const float* __restrict__ sgw, float* __restrict__ comb,
    float* __restrict__ sgate) {
  __shared__ float red[9][256];
  const int t = blockIdx.x;
  const int tid = threadIdx.x;
  float acc[9];
#pragma unroll
  for (int j = 0; j < 9; ++j) acc[j] = 0.f;
  const float* xr = x + (size_t)t * H_DIM;
  for (int h = tid; h < H_DIM; h += 256) {
    float xv = xr[h];
#pragma unroll
    for (int e = 0; e < E_EXP; ++e) acc[e] += xv * gw[e * H_DIM + h];
    acc[8] += xv * sgw[h];
  }
#pragma unroll
  for (int j = 0; j < 9; ++j) red[j][tid] = acc[j];
  __syncthreads();
  for (int s = 128; s > 0; s >>= 1) {
    if (tid < s) {
#pragma unroll
      for (int j = 0; j < 9; ++j) red[j][tid] += red[j][tid + s];
    }
    __syncthreads();
  }
  if (tid == 0) {
    float lg[8], p[8];
    float mx = -1e30f;
    for (int e = 0; e < E_EXP; ++e) { lg[e] = red[e][0]; mx = fmaxf(mx, lg[e]); }
    float sum = 0.f;
    for (int e = 0; e < E_EXP; ++e) { p[e] = __expf(lg[e] - mx); sum += p[e]; }
    float inv = 1.f / sum;
    for (int e = 0; e < E_EXP; ++e) p[e] *= inv;
    float cw[8], tmp[8];
    for (int e = 0; e < E_EXP; ++e) { cw[e] = 0.f; tmp[e] = p[e]; }
    float tsum = 0.f;
    for (int k = 0; k < TOPK; ++k) {
      int bi = 0; float bv = -1.f;
      for (int e = 0; e < E_EXP; ++e) if (tmp[e] > bv) { bv = tmp[e]; bi = e; }
      cw[bi] = bv; tsum += bv; tmp[bi] = -2.f;
    }
    float invt = 1.f / tsum;
    for (int e = 0; e < E_EXP; ++e) comb[(size_t)t * E_EXP + e] = cw[e] * invt;
    sgate[t] = 1.f / (1.f + __expf(-red[8][0]));
  }
}

// ---------------- Cast x (fp32) -> bf16 once.
__global__ __launch_bounds__(256) void cast_bf16_kernel(
    const float* __restrict__ x, unsigned short* __restrict__ xb, int n) {
  int i = (blockIdx.x * 256 + threadIdx.x) * 4;
  if (i >= n) return;
  float4 v = *(const float4*)(x + i);
  uint2 o; o.x = pack2(v.x, v.y); o.y = pack2(v.z, v.w);
  *(uint2*)(xb + i) = o;
}

// ---------------- Zero-init output accumulator.
__global__ __launch_bounds__(256) void zero_out_kernel(float* __restrict__ out, int n) {
  int i = (blockIdx.x * 256 + threadIdx.x) * 4;
  if (i >= n) return;
  float4 z; z.x = z.y = z.z = z.w = 0.f;
  *(float4*)(out + i) = z;
}

// ---------------- Stable per-expert token compaction (deterministic prefix scan).
__global__ __launch_bounds__(1024) void dispatch_kernel(
    const float* __restrict__ comb, int* __restrict__ tidx,
    float* __restrict__ tw, int* __restrict__ counts) {
  __shared__ int sc[T_TOK];
  const int e = blockIdx.x;
  const int t = threadIdx.x;
  float w = comb[(size_t)t * E_EXP + e];
  int f = (w > 0.f) ? 1 : 0;
  sc[t] = f;
  __syncthreads();
  for (int off = 1; off < T_TOK; off <<= 1) {
    int add = (t >= off) ? sc[t - off] : 0;
    __syncthreads();
    sc[t] += add;
    __syncthreads();
  }
  int total = sc[T_TOK - 1];
  int pos = sc[t] - f;
  if (f) { tidx[e * T_TOK + pos] = t; tw[e * T_TOK + pos] = w; }
  if (t >= total) { tidx[e * T_TOK + t] = 0; tw[e * T_TOK + t] = 0.f; }  // padding
  if (t == 0) counts[e] = total;
}

// ---------------- Dual GEMM: Act[pos,N] = silu(X@Wg^T) * (X@Wu^T); bf16 WMMA.
// A tile: async global->LDS (ASYNCcnt). B tiles: global->reg->pack->LDS.
// Double-buffered LDS, 1 barrier/iter.
__global__ __launch_bounds__(256) void moe_act_gemm(
    const unsigned short* __restrict__ Xb,  // [T,K] bf16
    const float* __restrict__ Wg,           // [N,K] fp32
    const float* __restrict__ Wu,           // [N,K] fp32
    unsigned short* __restrict__ Act,       // [rows,N] bf16 (compacted rows)
    const int* __restrict__ gather,         // [T] or null
    const int* __restrict__ cntp,           // or null
    int M, int N, int K) {
  const int cnt = cntp ? cntp[0] : M;
  const int bm = blockIdx.y * MT;
  if (bm >= cnt) return;
  const int bn = blockIdx.x * NT;

  __shared__ unsigned short As[2][MT * LDP];
  __shared__ unsigned short Bg[2][NT * LDP];
  __shared__ unsigned short Bu[2][NT * LDP];

  const int tid = threadIdx.x;
  const int lane = tid & 31;
  const int wave = tid >> 5;
  const int wm = wave >> 1, wn = wave & 1;
  const int hfl = lane >> 4, l16 = lane & 15;

  const int arow = tid >> 1, akh = (tid & 1) * 16;
  const int brow = tid >> 2, bseg = (tid & 3) * 8;
  const int grow = gather ? gather[bm + arow] : (bm + arow);

  const unsigned short* Ap = Xb + (size_t)grow * K + akh;
  const float* Gp = Wg + (size_t)(bn + brow) * K + bseg;
  const float* Up = Wu + (size_t)(bn + brow) * K + bseg;
  unsigned short* AsD0 = &As[0][arow * LDP + akh];
  unsigned short* AsD1 = &As[1][arow * LDP + akh];

  v8f accG[2][2], accU[2][2];
#pragma unroll
  for (int i = 0; i < 2; ++i)
#pragma unroll
    for (int j = 0; j < 2; ++j) { accG[i][j] = v8f_zero(); accU[i][j] = v8f_zero(); }

  // prologue: stage tile 0
  async_cp16(Ap, AsD0);
  async_cp16(Ap + 8, AsD0 + 8);
  float4 rg0 = *(const float4*)(Gp);
  float4 rg1 = *(const float4*)(Gp + 4);
  float4 ru0 = *(const float4*)(Up);
  float4 ru1 = *(const float4*)(Up + 4);
  *(uint4*)(&Bg[0][brow * LDP + bseg]) =
      make_uint4(pack2(rg0.x, rg0.y), pack2(rg0.z, rg0.w), pack2(rg1.x, rg1.y), pack2(rg1.z, rg1.w));
  *(uint4*)(&Bu[0][brow * LDP + bseg]) =
      make_uint4(pack2(ru0.x, ru0.y), pack2(ru0.z, ru0.w), pack2(ru1.x, ru1.y), pack2(ru1.z, ru1.w));
  async_fence();
  __syncthreads();

  const int iters = K / KT;
  for (int it = 0; it < iters; ++it) {
    const int cur = it & 1;
    const bool more = (it + 1 < iters);
    if (more) {  // next A tile: async copy straight into the other LDS buffer
      const int k = (it + 1) * KT;
      unsigned short* ad = cur ? AsD0 : AsD1;
      async_cp16(Ap + k, ad);
      async_cp16(Ap + k + 8, ad + 8);
      rg0 = *(const float4*)(Gp + k);
      rg1 = *(const float4*)(Gp + k + 4);
      ru0 = *(const float4*)(Up + k);
      ru1 = *(const float4*)(Up + k + 4);
      if (it + 2 < iters) {
        __builtin_prefetch(Gp + (it + 2) * KT, 0, 1);
        __builtin_prefetch(Up + (it + 2) * KT, 0, 1);
      }
    }

    v16bf af[2];
#pragma unroll
    for (int mt = 0; mt < 2; ++mt) {  // A 16x32: lane<16 K0..7|16..23, lane>=16 K8..15|24..31
      const int m = wm * 32 + mt * 16 + l16;
      v8bf lo = *(const v8bf*)(&As[cur][m * LDP + hfl * 8]);
      v8bf hi = *(const v8bf*)(&As[cur][m * LDP + 16 + hfl * 8]);
      v16bf a;
#pragma unroll
      for (int i = 0; i < 8; ++i) { a[i] = lo[i]; a[i + 8] = hi[i]; }
      af[mt] = a;
    }
#pragma unroll
    for (int nt = 0; nt < 2; ++nt) {  // B 32x16: lane<16 K0..7|16..23, lane>=16 K8..15|24..31
      const int n = wn * 32 + nt * 16 + l16;
      v8bf bgl = *(const v8bf*)(&Bg[cur][n * LDP + hfl * 8]);
      v8bf bgh = *(const v8bf*)(&Bg[cur][n * LDP + 16 + hfl * 8]);
      v8bf bul = *(const v8bf*)(&Bu[cur][n * LDP + hfl * 8]);
      v8bf buh = *(const v8bf*)(&Bu[cur][n * LDP + 16 + hfl * 8]);
      v16bf bg, bu;
#pragma unroll
      for (int i = 0; i < 8; ++i) { bg[i] = bgl[i]; bg[i + 8] = bgh[i]; bu[i] = bul[i]; bu[i + 8] = buh[i]; }
#pragma unroll
      for (int mt = 0; mt < 2; ++mt) {
        accG[mt][nt] = __builtin_amdgcn_wmma_f32_16x16x32_bf16(
            false, af[mt], false, bg, (short)0, accG[mt][nt], false, false);
        accU[mt][nt] = __builtin_amdgcn_wmma_f32_16x16x32_bf16(
            false, af[mt], false, bu, (short)0, accU[mt][nt], false, false);
      }
    }

    if (more) {  // convert+store next B stage (waits on the global loads issued above)
      const int nxt = cur ^ 1;
      *(uint4*)(&Bg[nxt][brow * LDP + bseg]) =
          make_uint4(pack2(rg0.x, rg0.y), pack2(rg0.z, rg0.w), pack2(rg1.x, rg1.y), pack2(rg1.z, rg1.w));
      *(uint4*)(&Bu[nxt][brow * LDP + bseg]) =
          make_uint4(pack2(ru0.x, ru0.y), pack2(ru0.z, ru0.w), pack2(ru1.x, ru1.y), pack2(ru1.z, ru1.w));
    }
    async_fence();
    __syncthreads();
  }

  // epilogue: silu(g)*u -> bf16, compacted row order
#pragma unroll
  for (int mt = 0; mt < 2; ++mt)
#pragma unroll
    for (int nt = 0; nt < 2; ++nt)
#pragma unroll
      for (int r = 0; r < 8; ++r) {
        const int row = bm + wm * 32 + mt * 16 + hfl * 8 + r;  // C layout: lanes>=16 -> M+8
        const int col = bn + wn * 32 + nt * 16 + l16;
        float gv = accG[mt][nt][r], uv = accU[mt][nt][r];
        float sv = gv / (1.f + __expf(-gv));
        Act[(size_t)row * N + col] = f2bf(sv * uv);
      }
}

// ---------------- Down-proj GEMM + per-row scale, scatter-accumulate into fp32 out.
__global__ __launch_bounds__(256) void moe_down_gemm(
    const unsigned short* __restrict__ Ab,  // [rows,K] bf16 (compacted)
    const float* __restrict__ Wd,           // [N,K] fp32
    float* __restrict__ Out,                // [T,N] fp32 accumulator
    const float* __restrict__ scale,        // per compacted row
    const int* __restrict__ scatter,        // [T] or null
    const int* __restrict__ cntp,           // or null
    int M, int N, int K) {
  const int cnt = cntp ? cntp[0] : M;
  const int bm = blockIdx.y * MT;
  if (bm >= cnt) return;
  const int bn = blockIdx.x * NT;

  __shared__ unsigned short As[2][MT * LDP];
  __shared__ unsigned short Bd[2][NT * LDP];

  const int tid = threadIdx.x;
  const int lane = tid & 31;
  const int wave = tid >> 5;
  const int wm = wave >> 1, wn = wave & 1;
  const int hfl = lane >> 4, l16 = lane & 15;

  const int arow = tid >> 1, akh = (tid & 1) * 16;
  const int brow = tid >> 2, bseg = (tid & 3) * 8;

  const unsigned short* Ap = Ab + (size_t)(bm + arow) * K + akh;
  const float* Dp = Wd + (size_t)(bn + brow) * K + bseg;
  unsigned short* AsD0 = &As[0][arow * LDP + akh];
  unsigned short* AsD1 = &As[1][arow * LDP + akh];

  v8f acc[2][2];
#pragma unroll
  for (int i = 0; i < 2; ++i)
#pragma unroll
    for (int j = 0; j < 2; ++j) acc[i][j] = v8f_zero();

  async_cp16(Ap, AsD0);
  async_cp16(Ap + 8, AsD0 + 8);
  float4 rd0 = *(const float4*)(Dp);
  float4 rd1 = *(const float4*)(Dp + 4);
  *(uint4*)(&Bd[0][brow * LDP + bseg]) =
      make_uint4(pack2(rd0.x, rd0.y), pack2(rd0.z, rd0.w), pack2(rd1.x, rd1.y), pack2(rd1.z, rd1.w));
  async_fence();
  __syncthreads();

  const int iters = K / KT;
  for (int it = 0; it < iters; ++it) {
    const int cur = it & 1;
    const bool more = (it + 1 < iters);
    if (more) {
      const int k = (it + 1) * KT;
      unsigned short* ad = cur ? AsD0 : AsD1;
      async_cp16(Ap + k, ad);
      async_cp16(Ap + k + 8, ad + 8);
      rd0 = *(const float4*)(Dp + k);
      rd1 = *(const float4*)(Dp + k + 4);
      if (it + 2 < iters) __builtin_prefetch(Dp + (it + 2) * KT, 0, 1);
    }

    v16bf af[2];
#pragma unroll
    for (int mt = 0; mt < 2; ++mt) {
      const int m = wm * 32 + mt * 16 + l16;
      v8bf lo = *(const v8bf*)(&As[cur][m * LDP + hfl * 8]);
      v8bf hi = *(const v8bf*)(&As[cur][m * LDP + 16 + hfl * 8]);
      v16bf a;
#pragma unroll
      for (int i = 0; i < 8; ++i) { a[i] = lo[i]; a[i + 8] = hi[i]; }
      af[mt] = a;
    }
#pragma unroll
    for (int nt = 0; nt < 2; ++nt) {
      const int n = wn * 32 + nt * 16 + l16;
      v8bf bl = *(const v8bf*)(&Bd[cur][n * LDP + hfl * 8]);
      v8bf bh = *(const v8bf*)(&Bd[cur][n * LDP + 16 + hfl * 8]);
      v16bf b;
#pragma unroll
      for (int i = 0; i < 8; ++i) { b[i] = bl[i]; b[i + 8] = bh[i]; }
#pragma unroll
      for (int mt = 0; mt < 2; ++mt)
        acc[mt][nt] = __builtin_amdgcn_wmma_f32_16x16x32_bf16(
            false, af[mt], false, b, (short)0, acc[mt][nt], false, false);
    }

    if (more) {
      const int nxt = cur ^ 1;
      *(uint4*)(&Bd[nxt][brow * LDP + bseg]) =
          make_uint4(pack2(rd0.x, rd0.y), pack2(rd0.z, rd0.w), pack2(rd1.x, rd1.y), pack2(rd1.z, rd1.w));
    }
    async_fence();
    __syncthreads();
  }

#pragma unroll
  for (int mt = 0; mt < 2; ++mt) {
    float scl[8];
    int orow[8];
    int valid[8];
#pragma unroll
    for (int r = 0; r < 8; ++r) {
      const int prow = bm + wm * 32 + mt * 16 + hfl * 8 + r;
      valid[r] = (prow < cnt);
      scl[r] = scale[prow];
      orow[r] = scatter ? scatter[prow] : prow;
    }
#pragma unroll
    for (int nt = 0; nt < 2; ++nt) {
      const int col = bn + wn * 32 + nt * 16 + l16;
#pragma unroll
      for (int r = 0; r < 8; ++r) {
        if (valid[r]) {
          size_t o = (size_t)orow[r] * N + col;
          Out[o] += scl[r] * acc[mt][nt][r];
        }
      }
    }
  }
}

extern "C" void kernel_launch(void* const* d_in, const int* in_sizes, int n_in,
                              void* d_out, int out_size, void* d_ws, size_t ws_size,
                              hipStream_t stream) {
  const float* x   = (const float*)d_in[0];  // [1,T,H]
  const float* gw  = (const float*)d_in[1];  // [E,H]
  const float* sgw = (const float*)d_in[2];  // [1,H]
  const float* wg  = (const float*)d_in[3];  // [E,I,H]
  const float* wu  = (const float*)d_in[4];  // [E,I,H]
  const float* wd  = (const float*)d_in[5];  // [E,H,I]
  const float* swg = (const float*)d_in[6];  // [IS,H]
  const float* swu = (const float*)d_in[7];  // [IS,H]
  const float* swd = (const float*)d_in[8];  // [H,IS]
  float* out = (float*)d_out;

  char* ws = (char*)d_ws;
  size_t off = 0;
  unsigned short* xb = (unsigned short*)(ws + off); off += (size_t)T_TOK * H_DIM * 2;
  float* comb  = (float*)(ws + off); off += (size_t)T_TOK * E_EXP * 4;
  float* sg    = (float*)(ws + off); off += (size_t)T_TOK * 4;
  int*   tidx  = (int*)(ws + off);   off += (size_t)E_EXP * T_TOK * 4;
  float* tw    = (float*)(ws + off); off += (size_t)E_EXP * T_TOK * 4;
  int* counts  = (int*)(ws + off);   off += 256;
  unsigned short* act = (unsigned short*)(ws + off);  // T*IS bf16, reused per expert

  router_kernel<<<T_TOK, 256, 0, stream>>>(x, gw, sgw, comb, sg);
  cast_bf16_kernel<<<(T_TOK * H_DIM) / 1024, 256, 0, stream>>>(x, xb, T_TOK * H_DIM);
  dispatch_kernel<<<E_EXP, T_TOK, 0, stream>>>(comb, tidx, tw, counts);
  zero_out_kernel<<<(T_TOK * H_DIM) / 1024, 256, 0, stream>>>(out, T_TOK * H_DIM);

  for (int e = 0; e < E_EXP; ++e) {
    moe_act_gemm<<<dim3(I_DIM / NT, T_TOK / MT), 256, 0, stream>>>(
        xb, wg + (size_t)e * I_DIM * H_DIM, wu + (size_t)e * I_DIM * H_DIM, act,
        tidx + e * T_TOK, counts + e, T_TOK, I_DIM, H_DIM);
    moe_down_gemm<<<dim3(H_DIM / NT, T_TOK / MT), 256, 0, stream>>>(
        act, wd + (size_t)e * H_DIM * I_DIM, out, tw + e * T_TOK,
        tidx + e * T_TOK, counts + e, T_TOK, H_DIM, I_DIM);
  }
  // shared expert: dense rows, sigmoid gate scale
  moe_act_gemm<<<dim3(IS_DIM / NT, T_TOK / MT), 256, 0, stream>>>(
      xb, swg, swu, act, nullptr, nullptr, T_TOK, IS_DIM, H_DIM);
  moe_down_gemm<<<dim3(H_DIM / NT, T_TOK / MT), 256, 0, stream>>>(
      act, swd, out, sg, nullptr, nullptr, T_TOK, H_DIM, IS_DIM);
}
